// SwinTransformerBlock_41197326303643
// MI455X (gfx1250) — matrix operand order repaired and checked by reading, not compiled, not run on hardware
//
#include <hip/hip_runtime.h>
#include <hip/hip_bf16.h>
#include <math.h>

// ---------------- constants (from reference) ----------------
#define BATCH   8
#define HDIM    128          // H = W = 128
#define Cdim    192
#define HEADS   6
#define HD      32           // head dim
#define WSZ     8
#define NTOK    64           // WSZ*WSZ
#define SHIFT   4
#define NWSIDE  16           // 128/8
#define NWIN    256          // per image
#define BWIN    2048         // BATCH * NWIN
#define ROWS    131072       // BWIN * NTOK
#define HIDDEN  768
#define QSCALE  0.17677669529663687f   // 32^-0.5

typedef __attribute__((ext_vector_type(16))) __bf16 v16bf;
typedef __attribute__((ext_vector_type(8)))  __bf16 v8bf;
typedef __attribute__((ext_vector_type(8)))  float  v8f;

// ---------------- WMMA helpers (ISA 7.12.2 layouts) ----------------
__device__ __forceinline__ v8f zero8() {
    v8f z;
#pragma unroll
    for (int i = 0; i < 8; ++i) z[i] = 0.0f;
    return z;
}

// A fragment: 16x32 bf16, rows row0..row0+15, K = k0..k0+31, src row-major ld halves.
__device__ __forceinline__ v16bf frag_a(const __bf16* base, int ld, int row0, int k0) {
    const int lane = threadIdx.x & 31;
    const int m = lane & 15, hi = lane >> 4;
    const __bf16* p = base + (size_t)(row0 + m) * ld + k0 + hi * 8;
    v8bf lo = *(const v8bf*)(p);
    v8bf hh = *(const v8bf*)(p + 16);
    v16bf r;
#pragma unroll
    for (int i = 0; i < 8; ++i) { r[i] = lo[i]; r[i + 8] = hh[i]; }
    return r;
}

// B fragment: 32x16 bf16, element (k,n) from N-major source: baseT[(n0+n)*ld + k].
__device__ __forceinline__ v16bf frag_b(const __bf16* baseT, int ld, int n0, int k0) {
    const int lane = threadIdx.x & 31;
    const int n = lane & 15, hi = lane >> 4;
    const __bf16* p = baseT + (size_t)(n0 + n) * ld + k0 + hi * 8;
    v8bf lo = *(const v8bf*)(p);
    v8bf hh = *(const v8bf*)(p + 16);
    v16bf r;
#pragma unroll
    for (int i = 0; i < 8; ++i) { r[i] = lo[i]; r[i + 8] = hh[i]; }
    return r;
}

__device__ __forceinline__ v8f wmma_bf16(v16bf a, v16bf b, v8f c) {
    return __builtin_amdgcn_wmma_f32_16x16x32_bf16(false, a, false, b, (short)0, c, false, false);
}

__device__ __forceinline__ float wave_red_sum(float v) {
#pragma unroll
    for (int o = 16; o > 0; o >>= 1) v += __shfl_xor(v, o, 32);
    return v;
}

// CDNA5 async global->LDS staging: each thread issues 1/nthreads of the `chunks16` 16-byte
// async copies (GLOBAL_LOAD_ASYNC_TO_LDS_B128, tracked by ASYNCcnt), then
// drains its own ASYNCcnt. Caller must __syncthreads() afterwards.
__device__ __forceinline__ void async_stage_tile(__bf16* lds_base,
                                                 const __bf16* gsrc,
                                                 int chunks16, int nthreads) {
    const unsigned lds0 = (unsigned)(uintptr_t)lds_base;   // low 32 bits = LDS addr
    for (int idx = threadIdx.x; idx < chunks16; idx += nthreads) {
        unsigned ldsa = lds0 + (unsigned)idx * 16u;
        const char* ga = (const char*)gsrc + (size_t)idx * 16u;
        asm volatile("global_load_async_to_lds_b128 %0, %1, off"
                     :: "v"(ldsa), "v"(ga) : "memory");
    }
    asm volatile("s_wait_asynccnt 0x0" ::: "memory");
}

// ---------------- K0: weight convert + transpose to bf16 N-major ----------------
__global__ void convert_weights_kernel(const float* __restrict__ qkv_w,
                                       const float* __restrict__ proj_w,
                                       const float* __restrict__ fc1_w,
                                       const float* __restrict__ fc2_w,
                                       __bf16* __restrict__ WqkvT,
                                       __bf16* __restrict__ WprojT,
                                       __bf16* __restrict__ W1T,
                                       __bf16* __restrict__ W2T) {
    int i = blockIdx.x * blockDim.x + threadIdx.x;
    if (i < 110592) {                         // qkv_w [192,576] -> [576,192]
        int o = i / 192, c = i % 192;
        WqkvT[i] = (__bf16)qkv_w[c * 576 + o];
    } else if (i < 147456) {                  // proj_w [192,192] -> T
        int j = i - 110592;
        int o = j / 192, c = j % 192;
        WprojT[j] = (__bf16)proj_w[c * 192 + o];
    } else if (i < 294912) {                  // fc1_w [192,768] -> [768,192]
        int j = i - 147456;
        int o = j / 192, c = j % 192;
        W1T[j] = (__bf16)fc1_w[c * 768 + o];
    } else if (i < 442368) {                  // fc2_w [768,192] -> [192,768]
        int j = i - 294912;
        int o = j / 768, h = j % 768;
        W2T[j] = (__bf16)fc2_w[h * 192 + o];
    }
}

// ---------------- K1: dynamic position-bias MLP -> pos[225][6] ----------------
__device__ __forceinline__ void lnrelu12(float* v, const float* g, const float* b) {
    float s = 0.0f;
#pragma unroll
    for (int d = 0; d < 12; ++d) s += v[d];
    float mu = s * (1.0f / 12.0f);
    float sq = 0.0f;
#pragma unroll
    for (int d = 0; d < 12; ++d) { float t = v[d] - mu; sq += t * t; }
    float rstd = rsqrtf(sq * (1.0f / 12.0f) + 1e-5f);
#pragma unroll
    for (int d = 0; d < 12; ++d) {
        float t = (v[d] - mu) * rstd * g[d] + b[d];
        v[d] = t > 0.0f ? t : 0.0f;
    }
}

__global__ void pos_mlp_kernel(const float* __restrict__ pp_w, const float* __restrict__ pp_b,
                               const float* __restrict__ p1_g, const float* __restrict__ p1_b,
                               const float* __restrict__ p1_w, const float* __restrict__ p1_bias,
                               const float* __restrict__ p2_g, const float* __restrict__ p2_b,
                               const float* __restrict__ p2_w, const float* __restrict__ p2_bias,
                               const float* __restrict__ p3_g, const float* __restrict__ p3_b,
                               const float* __restrict__ p3_w, const float* __restrict__ p3_bias,
                               float* __restrict__ pos) {
    int i = blockIdx.x * blockDim.x + threadIdx.x;
    if (i >= 225) return;
    float fh = (float)(i / 15) - 7.0f;
    float fw = (float)(i % 15) - 7.0f;
    float cur[12], nxt[12];
#pragma unroll
    for (int d = 0; d < 12; ++d) cur[d] = fh * pp_w[d] + fw * pp_w[12 + d] + pp_b[d];
    lnrelu12(cur, p1_g, p1_b);
#pragma unroll
    for (int j = 0; j < 12; ++j) {
        float s = 0.0f;
#pragma unroll
        for (int d = 0; d < 12; ++d) s += cur[d] * p1_w[d * 12 + j];
        nxt[j] = s + p1_bias[j];
    }
#pragma unroll
    for (int d = 0; d < 12; ++d) cur[d] = nxt[d];
    lnrelu12(cur, p2_g, p2_b);
#pragma unroll
    for (int j = 0; j < 12; ++j) {
        float s = 0.0f;
#pragma unroll
        for (int d = 0; d < 12; ++d) s += cur[d] * p2_w[d * 12 + j];
        nxt[j] = s + p2_bias[j];
    }
#pragma unroll
    for (int d = 0; d < 12; ++d) cur[d] = nxt[d];
    lnrelu12(cur, p3_g, p3_b);
#pragma unroll
    for (int h = 0; h < 6; ++h) {
        float s = 0.0f;
#pragma unroll
        for (int d = 0; d < 12; ++d) s += cur[d] * p3_w[d * 6 + h];
        pos[i * 6 + h] = s + p3_bias[h];
    }
}

// ---------------- K2: LN1 + roll(-4,-4) + window partition -> xw bf16 ----------------
__global__ void ln_partition_kernel(const float* __restrict__ x,
                                    const float* __restrict__ g, const float* __restrict__ b,
                                    __bf16* __restrict__ xw) {
    const int wave = threadIdx.x >> 5, lane = threadIdx.x & 31;
    const int row = blockIdx.x * 8 + wave;           // [0, ROWS)
    const int win = row >> 6, n = row & 63;
    const int bb = win >> 8, wi = win & 255;
    const int hh = (wi >> 4) * 8 + (n >> 3);
    const int wwp = (wi & 15) * 8 + (n & 7);
    const int hs = (hh + SHIFT) & 127;
    const int wsrc = (wwp + SHIFT) & 127;
    const float* src = x + ((size_t)bb * 16384 + hs * 128 + wsrc) * Cdim;

    float vals[6];
    float s = 0.0f, sq = 0.0f;
#pragma unroll
    for (int t = 0; t < 6; ++t) {
        float v = src[lane + 32 * t];
        vals[t] = v; s += v; sq += v * v;
    }
    s = wave_red_sum(s);
    sq = wave_red_sum(sq);
    float mean = s * (1.0f / 192.0f);
    float var = sq * (1.0f / 192.0f) - mean * mean;
    float rstd = rsqrtf(var + 1e-5f);

    __bf16* dst = xw + (size_t)row * Cdim;
#pragma unroll
    for (int t = 0; t < 6; ++t) {
        int c = lane + 32 * t;
        dst[c] = (__bf16)((vals[t] - mean) * rstd * g[c] + b[c]);
    }
}

// ---------------- K3: QKV GEMM (per window 64x576x192) ----------------
__global__ void qkv_gemm_kernel(const __bf16* __restrict__ xw,
                                const __bf16* __restrict__ WqkvT,
                                const float* __restrict__ qkv_b,
                                __bf16* __restrict__ qb, __bf16* __restrict__ kb,
                                __bf16* __restrict__ vb) {
    __shared__ __align__(16) __bf16 Atile[NTOK * Cdim];    // 24 KB
    const int win = blockIdx.x;
    async_stage_tile(Atile, xw + (size_t)win * NTOK * Cdim, 1536, 256);
    __syncthreads();

    const int wave = threadIdx.x >> 5, lane = threadIdx.x & 31;
    const int mt = wave & 3, grp = wave >> 2;
    const int n = lane & 15, hi = lane >> 4;

    v16bf a[6];
#pragma unroll
    for (int ks = 0; ks < 6; ++ks) a[ks] = frag_a(Atile, Cdim, mt * 16, ks * 32);

    for (int t = 0; t < 18; ++t) {
        const int nt = grp * 18 + t;
        // prefetch next tile's weight row for this lane
        if (t + 1 < 18)
            __builtin_prefetch(WqkvT + (size_t)((nt + 1) * 16 + n) * Cdim, 0, 1);
        // batched B-fragment load (one clause), then WMMA burst
        v16bf bf[6];
#pragma unroll
        for (int ks = 0; ks < 6; ++ks) bf[ks] = frag_b(WqkvT, Cdim, nt * 16, ks * 32);
        v8f acc = zero8();
#pragma unroll
        for (int ks = 0; ks < 6; ++ks) acc = wmma_bf16(a[ks], bf[ks], acc);

        const int col = nt * 16 + n;                 // [0,576)
        const int mat = col / 192, rem = col % 192;
        const int head = rem >> 5, dd = rem & 31;
        __bf16* dst = (mat == 0) ? qb : ((mat == 1) ? kb : vb);
        const float bias = qkv_b[col];
        const float scl = (mat == 0) ? QSCALE : 1.0f;
        const size_t base = (((size_t)win * HEADS + head) * NTOK) * HD + dd;
#pragma unroll
        for (int i = 0; i < 8; ++i) {
            int tok = mt * 16 + hi * 8 + i;
            dst[base + (size_t)tok * HD] = (__bf16)((acc[i] + bias) * scl);
        }
    }
}

// ---------------- K4: attention per (window, head) ----------------
__global__ void attn_kernel(const __bf16* __restrict__ qb, const __bf16* __restrict__ kb,
                            const __bf16* __restrict__ vb, const float* __restrict__ pos,
                            __bf16* __restrict__ attn_out) {
    __shared__ __align__(16) float  S[NTOK * NTOK];    // 16 KB
    __shared__ __align__(16) __bf16 P[NTOK * NTOK];    //  8 KB
    __shared__ __align__(16) __bf16 VT[HD * NTOK];     //  4 KB (transposed V)
    __shared__ float posl[225 * 6];
    __shared__ int ids[NTOK];

    const int bid = blockIdx.x;
    const int win = bid / HEADS, head = bid % HEADS;
    const int wi = win & 255;
    const size_t hb = (((size_t)win * HEADS + head) * NTOK) * HD;
    const __bf16* q = qb + hb;
    const __bf16* k = kb + hb;
    const __bf16* v = vb + hb;

    // stage V transposed
    for (int idx = threadIdx.x; idx < NTOK * HD; idx += 128) {
        int tok = idx >> 5, dd = idx & 31;
        VT[dd * NTOK + tok] = v[idx];
    }
    for (int idx = threadIdx.x; idx < 225 * 6; idx += 128) posl[idx] = pos[idx];
    if (threadIdx.x < NTOK) {
        int nn = threadIdx.x;
        int hh = (wi >> 4) * 8 + (nn >> 3);
        int ww2 = (wi & 15) * 8 + (nn & 7);
        int hr = hh < 120 ? 0 : (hh < 124 ? 1 : 2);
        int wr = ww2 < 120 ? 0 : (ww2 < 124 ? 1 : 2);
        ids[nn] = hr * 3 + wr;
    }
    __syncthreads();

    const int wave = threadIdx.x >> 5, lane = threadIdx.x & 31;
    const int n = lane & 15, hi = lane >> 4;

    // S = q @ k^T  (K = 32 -> 1 WMMA per 16x16 tile); batch all 4 B-frags
    v16bf aq = frag_a(q, HD, wave * 16, 0);
    v16bf bk[4];
#pragma unroll
    for (int nt = 0; nt < 4; ++nt) bk[nt] = frag_b(k, HD, nt * 16, 0);
#pragma unroll
    for (int nt = 0; nt < 4; ++nt) {
        v8f acc = zero8();
        acc = wmma_bf16(aq, bk[nt], acc);
#pragma unroll
        for (int i = 0; i < 8; ++i)
            S[(wave * 16 + hi * 8 + i) * NTOK + nt * 16 + n] = acc[i];
    }
    __syncthreads();

    // softmax rows (bias + mask computed on the fly); threads 0..63 each own a row
    if (threadIdx.x < NTOK) {
        const int r = threadIdx.x;
        const int idr = ids[r];
        const int ri = r >> 3, rj = r & 7;
        float mx = -1e30f;
        for (int m = 0; m < NTOK; ++m) {
            int mi = m >> 3, mj = m & 7;
            int pidx = (ri - mi + 7) * 15 + (rj - mj + 7);
            float val = S[r * NTOK + m] + posl[pidx * 6 + head]
                      + ((idr != ids[m]) ? -100.0f : 0.0f);
            S[r * NTOK + m] = val;
            mx = fmaxf(mx, val);
        }
        float sum = 0.0f;
        for (int m = 0; m < NTOK; ++m) {
            float e = expf(S[r * NTOK + m] - mx);
            sum += e;
            S[r * NTOK + m] = e;
        }
        float inv = 1.0f / sum;
        for (int m = 0; m < NTOK; ++m)
            P[r * NTOK + m] = (__bf16)(S[r * NTOK + m] * inv);
    }
    __syncthreads();

    // O = P @ V ;  write into [row][192] with head column offset
    __bf16* outp = attn_out + ((size_t)win * NTOK) * Cdim + head * HD;
#pragma unroll
    for (int nt = 0; nt < 2; ++nt) {
        v16bf ap[2], bv[2];
#pragma unroll
        for (int ks = 0; ks < 2; ++ks) {
            ap[ks] = frag_a(P, NTOK, wave * 16, ks * 32);
            bv[ks] = frag_b(VT, NTOK, nt * 16, ks * 32);
        }
        v8f acc = zero8();
#pragma unroll
        for (int ks = 0; ks < 2; ++ks) acc = wmma_bf16(ap[ks], bv[ks], acc);
#pragma unroll
        for (int i = 0; i < 8; ++i)
            outp[(size_t)(wave * 16 + hi * 8 + i) * Cdim + nt * 16 + n] = (__bf16)acc[i];
    }
}

// ---------------- K5: proj GEMM + window reverse + roll(+4,+4) + shortcut ----------------
__global__ void proj_kernel(const __bf16* __restrict__ attn_out,
                            const __bf16* __restrict__ WprojT,
                            const float* __restrict__ proj_b,
                            const float* __restrict__ x, float* __restrict__ y) {
    __shared__ __align__(16) __bf16 Atile[NTOK * Cdim];
    const int win = blockIdx.x;
    async_stage_tile(Atile, attn_out + (size_t)win * NTOK * Cdim, 1536, 256);
    __syncthreads();

    const int wave = threadIdx.x >> 5, lane = threadIdx.x & 31;
    const int mt = wave & 3, grp = wave >> 2;
    const int bb = win >> 8, wi = win & 255;
    const int n = lane & 15, hi = lane >> 4;

    v16bf a[6];
#pragma unroll
    for (int ks = 0; ks < 6; ++ks) a[ks] = frag_a(Atile, Cdim, mt * 16, ks * 32);

#pragma unroll
    for (int t = 0; t < 6; ++t) {
        const int nt = grp * 6 + t;
        v16bf bf[6];
#pragma unroll
        for (int ks = 0; ks < 6; ++ks) bf[ks] = frag_b(WprojT, Cdim, nt * 16, ks * 32);
        v8f acc = zero8();
#pragma unroll
        for (int ks = 0; ks < 6; ++ks) acc = wmma_bf16(a[ks], bf[ks], acc);

        const int col = nt * 16 + n;
        const float bias = proj_b[col];
#pragma unroll
        for (int i = 0; i < 8; ++i) {
            int tok = mt * 16 + hi * 8 + i;
            int hh = (wi >> 4) * 8 + (tok >> 3);
            int ww2 = (wi & 15) * 8 + (tok & 7);
            int hf = (hh + SHIFT) & 127;
            int wf = (ww2 + SHIFT) & 127;
            size_t off = (((size_t)bb * 16384) + hf * 128 + wf) * Cdim + col;
            y[off] = x[off] + acc[i] + bias;
        }
    }
}

// ---------------- K6: fused MLP  out = y + (gelu(LN(y)@W1+b1)@W2+b2) ----------------
__global__ void mlp_kernel(const float* __restrict__ y,
                           const float* __restrict__ n2_g, const float* __restrict__ n2_b,
                           const __bf16* __restrict__ W1T, const float* __restrict__ fc1_b,
                           const __bf16* __restrict__ W2T, const float* __restrict__ fc2_b,
                           float* __restrict__ out) {
    __shared__ __align__(16) __bf16 Hs[NTOK * Cdim];   // 24 KB  (LN(y) in bf16)
    __shared__ __align__(16) __bf16 G[NTOK * 128];     // 16 KB  (gelu chunk)

    const int blk = blockIdx.x;
    const int wave = threadIdx.x >> 5, lane = threadIdx.x & 31;
    const size_t r0 = (size_t)blk * NTOK;

    // LN2 : one wave per row, 8 rows each
#pragma unroll
    for (int p = 0; p < 8; ++p) {
        const int rloc = wave * 8 + p;
        const float* src = y + (r0 + rloc) * Cdim;
        float vals[6];
        float s = 0.0f, sq = 0.0f;
#pragma unroll
        for (int t = 0; t < 6; ++t) {
            float v = src[lane + 32 * t];
            vals[t] = v; s += v; sq += v * v;
        }
        s = wave_red_sum(s);
        sq = wave_red_sum(sq);
        float mean = s * (1.0f / 192.0f);
        float var = sq * (1.0f / 192.0f) - mean * mean;
        float rstd = rsqrtf(var + 1e-5f);
#pragma unroll
        for (int t = 0; t < 6; ++t) {
            int c = lane + 32 * t;
            Hs[rloc * Cdim + c] = (__bf16)((vals[t] - mean) * rstd * n2_g[c] + n2_b[c]);
        }
    }
    __syncthreads();

    const int mt = wave & 3, half = wave >> 2;
    const int n = lane & 15, hi = lane >> 4;

    v16bf ah[6];
#pragma unroll
    for (int ks = 0; ks < 6; ++ks) ah[ks] = frag_a(Hs, Cdim, mt * 16, ks * 32);

    v8f acc[6];
#pragma unroll
    for (int t = 0; t < 6; ++t) acc[t] = zero8();

    for (int hc = 0; hc < 6; ++hc) {           // 6 chunks of 128 hidden units
        // fc1 chunk + GELU -> G
#pragma unroll
        for (int t = 0; t < 4; ++t) {
            const int ntl = half * 4 + t;      // local n-tile within chunk [0,8)
            v16bf bf[6];
#pragma unroll
            for (int ks = 0; ks < 6; ++ks)
                bf[ks] = frag_b(W1T, Cdim, hc * 128 + ntl * 16, ks * 32);
            v8f a2 = zero8();
#pragma unroll
            for (int ks = 0; ks < 6; ++ks) a2 = wmma_bf16(ah[ks], bf[ks], a2);
            const int colh = hc * 128 + ntl * 16 + n;
            const float b1 = fc1_b[colh];
#pragma unroll
            for (int i = 0; i < 8; ++i) {
                float xg = a2[i] + b1;
                float gel = 0.5f * xg * (1.0f + erff(xg * 0.70710678118654752f));
                G[(mt * 16 + hi * 8 + i) * 128 + ntl * 16 + n] = (__bf16)gel;
            }
        }
        __syncthreads();

        // fc2 accumulate
        v16bf ag[4];
#pragma unroll
        for (int ks = 0; ks < 4; ++ks) ag[ks] = frag_a(G, 128, mt * 16, ks * 32);
#pragma unroll
        for (int t = 0; t < 6; ++t) {
            const int nt = half * 6 + t;
            v16bf bw[4];
#pragma unroll
            for (int ks = 0; ks < 4; ++ks)
                bw[ks] = frag_b(W2T, HIDDEN, nt * 16, hc * 128 + ks * 32);
#pragma unroll
            for (int ks = 0; ks < 4; ++ks) acc[t] = wmma_bf16(ag[ks], bw[ks], acc[t]);
        }
        __syncthreads();
    }

    // out = y + mlp + b2
#pragma unroll
    for (int t = 0; t < 6; ++t) {
        const int col = (half * 6 + t) * 16 + n;
        const float b2 = fc2_b[col];
#pragma unroll
        for (int i = 0; i < 8; ++i) {
            size_t row = r0 + mt * 16 + hi * 8 + i;
            out[row * Cdim + col] = y[row * Cdim + col] + acc[t][i] + b2;
        }
    }
}

// ---------------- host launch ----------------
extern "C" void kernel_launch(void* const* d_in, const int* in_sizes, int n_in,
                              void* d_out, int out_size, void* d_ws, size_t ws_size,
                              hipStream_t stream) {
    const float* x      = (const float*)d_in[0];
    const float* n1_g   = (const float*)d_in[4];
    const float* n1_b   = (const float*)d_in[5];
    const float* qkv_w  = (const float*)d_in[6];
    const float* qkv_b  = (const float*)d_in[7];
    const float* proj_w = (const float*)d_in[8];
    const float* proj_b = (const float*)d_in[9];
    const float* pp_w   = (const float*)d_in[10];
    const float* pp_b   = (const float*)d_in[11];
    const float* p1_g   = (const float*)d_in[12];
    const float* p1_b   = (const float*)d_in[13];
    const float* p1_w   = (const float*)d_in[14];
    const float* p1_bi  = (const float*)d_in[15];
    const float* p2_g   = (const float*)d_in[16];
    const float* p2_b   = (const float*)d_in[17];
    const float* p2_w   = (const float*)d_in[18];
    const float* p2_bi  = (const float*)d_in[19];
    const float* p3_g   = (const float*)d_in[20];
    const float* p3_b   = (const float*)d_in[21];
    const float* p3_w   = (const float*)d_in[22];
    const float* p3_bi  = (const float*)d_in[23];
    const float* n2_g   = (const float*)d_in[24];
    const float* n2_b   = (const float*)d_in[25];
    const float* fc1_w  = (const float*)d_in[26];
    const float* fc1_b  = (const float*)d_in[27];
    const float* fc2_w  = (const float*)d_in[28];
    const float* fc2_b  = (const float*)d_in[29];
    float* out = (float*)d_out;

    // ---- workspace layout (bf16 intermediates; aliased to minimize footprint) ----
    char* ws = (char*)d_ws;
    float*  pos    = (float*)ws;                              // 5400 B
    __bf16* WqkvT  = (__bf16*)(ws + 8192);                    // 221184 B
    __bf16* WprojT = WqkvT + 110592;                          // 73728 B
    __bf16* W1T    = WprojT + 36864;                          // 294912 B
    __bf16* W2T    = W1T + 147456;                            // 294912 B
    __bf16* xw     = (__bf16*)(ws + (1 << 20));               // 48 MB (ROWS*192)
    __bf16* attn_o = xw;                                      // alias: xw dead after qkv gemm
    __bf16* qb     = (__bf16*)(ws + (1 << 20) + (size_t)50331648);
    __bf16* kb     = qb + (size_t)ROWS * Cdim;
    __bf16* vb     = kb + (size_t)ROWS * Cdim;
    float*  ybuf   = (float*)qb;                              // alias: qkv dead after attention

    convert_weights_kernel<<<1728, 256, 0, stream>>>(qkv_w, proj_w, fc1_w, fc2_w,
                                                     WqkvT, WprojT, W1T, W2T);
    pos_mlp_kernel<<<1, 256, 0, stream>>>(pp_w, pp_b, p1_g, p1_b, p1_w, p1_bi,
                                          p2_g, p2_b, p2_w, p2_bi,
                                          p3_g, p3_b, p3_w, p3_bi, pos);
    ln_partition_kernel<<<ROWS / 8, 256, 0, stream>>>(x, n1_g, n1_b, xw);
    qkv_gemm_kernel<<<BWIN, 256, 0, stream>>>(xw, WqkvT, qkv_b, qb, kb, vb);
    attn_kernel<<<BWIN * HEADS, 128, 0, stream>>>(qb, kb, vb, pos, attn_o);
    proj_kernel<<<BWIN, 256, 0, stream>>>(attn_o, WprojT, proj_b, x, ybuf);
    mlp_kernel<<<BWIN, 256, 0, stream>>>(ybuf, n2_g, n2_b, W1T, fc1_b, W2T, fc2_b, out);
}
